// GNN_56401510531191
// MI455X (gfx1250) — compile-verified, matrix operand order checked by reading, not run on hardware
//
#include <hip/hip_runtime.h>
#include <hip/hip_bf16.h>

typedef _Float16 v16h __attribute__((ext_vector_type(16)));
typedef float    v8f  __attribute__((ext_vector_type(8)));

// exact parameter types for the gfx1250 async-to-LDS builtin
// (clang prints AS1 as __device__, AS3 as __shared__ in HIP diagnostics)
typedef int async_v4i __attribute__((vector_size(16)));
typedef __attribute__((address_space(1))) async_v4i* async_gptr;
typedef __attribute__((address_space(3))) async_v4i* async_lptr;

#define HDIM 32
#define BN_EPS 1e-5f
#define TILES_PER_WAVE 4

// ---------------------------------------------------------------------------
// utility: zero a float buffer
// ---------------------------------------------------------------------------
__global__ void gnn_zero_f32(float* __restrict__ p, int n) {
  int i = blockIdx.x * blockDim.x + threadIdx.x;
  if (i < n) p[i] = 0.0f;
}

// ---------------------------------------------------------------------------
// Pre-pack nn_W (+ nn_b as chunk 32) into f16 WMMA B-fragments.
// Layout: [chunk kk 0..32][nhalf 0..1][lane 0..31][elem 0..15]
// B_chunk[i,o] = nn_W[kk, i*32+o] (kk<32)  |  nn_b[i*32+o] (kk==32)
// Fragment slot mapping (16-bit operand): lane group g=lane>>4, col=lane&15,
//   slot (v,j): i = v<4 ? 8g+2v+j : 16+8g+2(v-4)+j ; o = nhalf*16+col
// ---------------------------------------------------------------------------
__global__ void gnn_pack_B(const float* __restrict__ nnW,
                           const float* __restrict__ nnb,
                           _Float16* __restrict__ out) {
  const int total = 33 * 2 * 32 * 16;
  int idx = blockIdx.x * blockDim.x + threadIdx.x;
  if (idx >= total) return;
  int elem = idx & 15;
  int lane = (idx >> 4) & 31;
  int nh   = (idx >> 9) & 1;
  int kk   = idx >> 10;               // 0..32
  int g = lane >> 4, col = lane & 15;
  int v = elem >> 1, j = elem & 1;
  int i = (v < 4) ? (g * 8 + 2 * v + j) : (16 + g * 8 + 2 * (v - 4) + j);
  int o = nh * 16 + col;
  float val = (kk < 32) ? nnW[kk * 1024 + i * 32 + o] : nnb[i * 32 + o];
  out[idx] = (_Float16)val;
}

// ---------------------------------------------------------------------------
// h = x @ node_W + node_b   (N x 30) @ (30 x 32)
// ---------------------------------------------------------------------------
__global__ void gnn_node_embed(const float* __restrict__ x,
                               const float* __restrict__ W,
                               const float* __restrict__ b,
                               float* __restrict__ h, int Nn) {
  int idx = blockIdx.x * blockDim.x + threadIdx.x;
  if (idx >= Nn * HDIM) return;
  int n = idx >> 5, c = idx & 31;
  const float* xr = x + (size_t)n * 30;
  float acc = b[c];
#pragma unroll
  for (int k = 0; k < 30; ++k) acc = fmaf(xr[k], W[k * HDIM + c], acc);
  h[idx] = acc;
}

// ---------------------------------------------------------------------------
// e = edge_attr @ edge_W + edge_b   (E x 6) @ (6 x 32)
// ---------------------------------------------------------------------------
__global__ void gnn_edge_embed(const float* __restrict__ ea,
                               const float* __restrict__ W,
                               const float* __restrict__ b,
                               float* __restrict__ e, int En) {
  int idx = blockIdx.x * blockDim.x + threadIdx.x;
  if (idx >= En * HDIM) return;
  int n = idx >> 5, c = idx & 31;
  const float* xr = ea + (size_t)n * 6;
  float acc = b[c];
#pragma unroll
  for (int k = 0; k < 6; ++k) acc = fmaf(xr[k], W[k * HDIM + c], acc);
  e[idx] = acc;
}

// ---------------------------------------------------------------------------
// Fused edge message + scatter-add using WMMA.
// One wave handles 16 edges per tile, TILES_PER_WAVE tiles.
// The 32 main B-chunks (64 KB) are staged in LDS once per block via
// async-to-LDS; bias chunk (kk==32, A = hf unscaled) comes from global once.
// Inner loop is software-pipelined: B fragments for chunk kk+1 are
// prefetched from LDS before chunk kk's WMMAs issue.
// ---------------------------------------------------------------------------
__global__ void gnn_edge_msg_wmma(const float* __restrict__ h,
                                  const float* __restrict__ e,
                                  const int* __restrict__ srcI,
                                  const int* __restrict__ dstI,
                                  const _Float16* __restrict__ packB,
                                  float* __restrict__ agg, int En, int nTiles) {
  __shared__ __align__(16) _Float16 ldsB[32 * 2 * 32 * 16];   // 64 KB

  // ---- cooperative staging of the 32 main chunks into LDS ----
  {
    const float4* gsrc = (const float4*)packB;
    float4* ldst = (float4*)ldsB;
#if defined(__gfx1250__) && __has_builtin(__builtin_amdgcn_global_load_async_to_lds_b128) && __has_builtin(__builtin_amdgcn_s_wait_asynccnt)
    for (int i = threadIdx.x; i < 4096; i += 256) {
      __builtin_amdgcn_global_load_async_to_lds_b128(
          (async_gptr)(gsrc + i),
          (async_lptr)(ldst + i),
          /*offset=*/0, /*cpol=*/0);
    }
    __builtin_amdgcn_s_wait_asynccnt(0);
#else
    for (int i = threadIdx.x; i < 4096; i += 256) ldst[i] = gsrc[i];
#endif
  }
  __syncthreads();

  const int lane = threadIdx.x & 31;
  const int waveId = blockIdx.x * 8 + (threadIdx.x >> 5);
  const int g   = lane >> 4;            // lane half
  const int row = lane & 15;            // A-matrix row handled by this lane

  const v16h* lp = (const v16h*)ldsB;                 // LDS fragments
  const v16h* gp = (const v16h*)packB;                // global (bias chunk)
  const v16h bbias0 = gp[(32 * 2 + 0) * 32 + lane];
  const v16h bbias1 = gp[(32 * 2 + 1) * 32 + lane];

#pragma unroll 1
  for (int t = 0; t < TILES_PER_WAVE; ++t) {
    int tile = waveId * TILES_PER_WAVE + t;
    if (tile >= nTiles) break;          // uniform per wave
    int base = tile * 16;
    int edge = base + row;
    if (edge >= En) edge = En - 1;      // clamp (compute garbage, skip scatter)

    // this lane's full e-row: chunk scales, converted to f16 up front
    _Float16 erh[32];
    const float4* ep = (const float4*)(e + (size_t)edge * HDIM);
#pragma unroll
    for (int q = 0; q < 8; ++q) {
      float4 tt = ep[q];
      erh[4 * q + 0] = (_Float16)tt.x; erh[4 * q + 1] = (_Float16)tt.y;
      erh[4 * q + 2] = (_Float16)tt.z; erh[4 * q + 3] = (_Float16)tt.w;
    }

    // h[src] as f16 A-fragment (documented 16-bit A slot order)
    int s = srcI[edge];
    const float* hp = h + (size_t)s * HDIM;
    v16h hf;
#pragma unroll
    for (int v = 0; v < 8; ++v) {
      int i0 = (v < 4) ? (g * 8 + 2 * v) : (16 + g * 8 + 2 * (v - 4));
      hf[2 * v + 0] = (_Float16)hp[i0 + 0];
      hf[2 * v + 1] = (_Float16)hp[i0 + 1];
    }

    // prime the B pipeline with chunk 0
    v16h b0 = lp[0 * 32 + lane];
    v16h b1 = lp[1 * 32 + lane];

    // bias chunk first: A = hf, no rescale needed (overlaps with ds loads)
    v8f c0 = {}; v8f c1 = {};
    c0 = __builtin_amdgcn_wmma_f32_16x16x32_f16(false, hf, false, bbias0,
                                                (short)0, c0, false, false);
    c1 = __builtin_amdgcn_wmma_f32_16x16x32_f16(false, hf, false, bbias1,
                                                (short)0, c1, false, false);

    // 32 main chunks, fully unrolled, B prefetched one chunk ahead
#pragma unroll
    for (int kk = 0; kk < 32; ++kk) {
      v16h nb0, nb1;
      if (kk < 31) {
        nb0 = lp[((kk + 1) * 2 + 0) * 32 + lane];
        nb1 = lp[((kk + 1) * 2 + 1) * 32 + lane];
      }
      v16h a = hf * erh[kk];            // 8x v_pk_mul_f16: per-row rescale
      c0 = __builtin_amdgcn_wmma_f32_16x16x32_f16(false, a, false, b0,
                                                  (short)0, c0, false, false);
      c1 = __builtin_amdgcn_wmma_f32_16x16x32_f16(false, a, false, b1,
                                                  (short)0, c1, false, false);
      b0 = nb0; b1 = nb1;
    }

    // scatter: D layout — lanes 0-15 hold M=r in vgpr r, lanes 16-31 M=8+r
    if (base + 16 <= En) {              // fast path: full tile, no guards
#pragma unroll
      for (int r = 0; r < 8; ++r) {
        int d = dstI[base + g * 8 + r];
        atomicAdd(&agg[(size_t)d * HDIM + row],      c0[r]);
        atomicAdd(&agg[(size_t)d * HDIM + 16 + row], c1[r]);
      }
    } else {
#pragma unroll
      for (int r = 0; r < 8; ++r) {
        int eidx = base + g * 8 + r;
        if (eidx < En) {
          int d = dstI[eidx];
          atomicAdd(&agg[(size_t)d * HDIM + row],      c0[r]);
          atomicAdd(&agg[(size_t)d * HDIM + 16 + row], c1[r]);
        }
      }
    }
  }
}

// ---------------------------------------------------------------------------
// t = agg + h @ root + cb (in-place into agg) + hierarchical BN statistics
// ---------------------------------------------------------------------------
__global__ void gnn_root_bnstat(const float* __restrict__ h,
                                const float* __restrict__ root,
                                const float* __restrict__ cb,
                                float* __restrict__ agg,
                                float* __restrict__ sums, int Nn) {
  __shared__ float s0[256];
  __shared__ float s1[256];
  int tid = threadIdx.x;
  int idx = blockIdx.x * 256 + tid;
  int c = tid & 31;
  float t = 0.0f;
  if (idx < Nn * HDIM) {
    int n = idx >> 5;
    const float* hr = h + (size_t)n * HDIM;
    float acc = cb[c] + agg[idx];
#pragma unroll
    for (int k = 0; k < HDIM; ++k) acc = fmaf(hr[k], root[k * HDIM + c], acc);
    agg[idx] = acc;
    t = acc;
  }
  s0[tid] = t;
  s1[tid] = t * t;
  __syncthreads();
  for (int off = 128; off >= 32; off >>= 1) {
    if (tid < off) { s0[tid] += s0[tid + off]; s1[tid] += s1[tid + off]; }
    __syncthreads();
  }
  if (tid < 32) {
    atomicAdd(&sums[c],      s0[tid]);
    atomicAdd(&sums[32 + c], s1[tid]);
  }
}

// ---------------------------------------------------------------------------
// h = relu((t - mean) * rsqrt(var + eps) * gamma + beta)
// ---------------------------------------------------------------------------
__global__ void gnn_bn_relu(const float* __restrict__ t,
                            const float* __restrict__ sums,
                            const float* __restrict__ gam,
                            const float* __restrict__ bet,
                            float* __restrict__ hout, int Nn) {
  int idx = blockIdx.x * blockDim.x + threadIdx.x;
  if (idx >= Nn * HDIM) return;
  int c = idx & 31;
  float invN = 1.0f / (float)Nn;
  float m  = sums[c] * invN;
  float vv = sums[32 + c] * invN - m * m;
  float r  = rsqrtf(vv + BN_EPS);
  float o  = (t[idx] - m) * r * gam[c] + bet[c];
  hout[idx] = o > 0.0f ? o : 0.0f;
}

// ---------------------------------------------------------------------------
// global mean pool: accumulate sums + counts per graph
// ---------------------------------------------------------------------------
__global__ void gnn_pool(const float* __restrict__ h,
                         const int* __restrict__ batch,
                         float* __restrict__ gx,
                         float* __restrict__ cnt, int Nn) {
  int idx = blockIdx.x * blockDim.x + threadIdx.x;
  if (idx >= Nn * HDIM) return;
  int n = idx >> 5, c = idx & 31;
  int gID = batch[n];
  atomicAdd(&gx[(size_t)gID * HDIM + c], h[idx]);
  if (c == 0) atomicAdd(&cnt[gID], 1.0f);
}

// ---------------------------------------------------------------------------
// pred = relu(gx/cnt @ lin1_W + b1) @ lin2_W + b2
// ---------------------------------------------------------------------------
__global__ void gnn_head(const float* __restrict__ gx,
                         const float* __restrict__ cnt,
                         const float* __restrict__ W1,
                         const float* __restrict__ b1,
                         const float* __restrict__ W2,
                         const float* __restrict__ b2,
                         float* __restrict__ out, int Gn) {
  int gid = blockIdx.x * blockDim.x + threadIdx.x;
  if (gid >= Gn) return;
  float inv = 1.0f / fmaxf(cnt[gid], 1.0f);
  float xr[HDIM];
#pragma unroll
  for (int c = 0; c < HDIM; ++c) xr[c] = gx[(size_t)gid * HDIM + c] * inv;
  float acc2 = b2[0];
#pragma unroll
  for (int j = 0; j < 16; ++j) {
    float a = b1[j];
#pragma unroll
    for (int k = 0; k < HDIM; ++k) a = fmaf(xr[k], W1[k * 16 + j], a);
    a = a > 0.0f ? a : 0.0f;
    acc2 = fmaf(a, W2[j], acc2);
  }
  out[gid] = acc2;
}

// ---------------------------------------------------------------------------
extern "C" void kernel_launch(void* const* d_in, const int* in_sizes, int n_in,
                              void* d_out, int out_size, void* d_ws, size_t ws_size,
                              hipStream_t stream) {
  (void)n_in; (void)ws_size;
  const float* x          = (const float*)d_in[0];
  const float* edge_attr  = (const float*)d_in[1];
  const int*   edge_index = (const int*)d_in[2];
  const int*   batch      = (const int*)d_in[3];
  const float* node_W = (const float*)d_in[4];
  const float* node_b = (const float*)d_in[5];
  const float* edge_W = (const float*)d_in[6];
  const float* edge_b = (const float*)d_in[7];
  const float* nn_W0 = (const float*)d_in[8];
  const float* nn_b0 = (const float*)d_in[9];
  const float* root0 = (const float*)d_in[10];
  const float* cb0   = (const float*)d_in[11];
  const float* g0    = (const float*)d_in[12];
  const float* be0   = (const float*)d_in[13];
  const float* nn_W1 = (const float*)d_in[14];
  const float* nn_b1 = (const float*)d_in[15];
  const float* root1 = (const float*)d_in[16];
  const float* cb1   = (const float*)d_in[17];
  const float* g1    = (const float*)d_in[18];
  const float* be1   = (const float*)d_in[19];
  const float* lin1_W = (const float*)d_in[20];
  const float* lin1_b = (const float*)d_in[21];
  const float* lin2_W = (const float*)d_in[22];
  const float* lin2_b = (const float*)d_in[23];

  const int N = in_sizes[0] / 30;
  const int E = in_sizes[1] / 6;
  const int G = out_size;
  const int* srcI = edge_index;
  const int* dstI = edge_index + E;

  // workspace carve-out (256B aligned blocks)
  char* ws = (char*)d_ws;
  size_t off = 0;
  auto carve = [&](size_t bytes) -> void* {
    off = (off + 255) & ~(size_t)255;
    void* p = ws + off;
    off += bytes;
    return p;
  };
  float*    hbuf   = (float*)carve((size_t)N * HDIM * sizeof(float));
  float*    ebuf   = (float*)carve((size_t)E * HDIM * sizeof(float));
  float*    agg    = (float*)carve((size_t)N * HDIM * sizeof(float));
  _Float16* packB0 = (_Float16*)carve((size_t)33 * 2 * 32 * 16 * sizeof(_Float16));
  _Float16* packB1 = (_Float16*)carve((size_t)33 * 2 * 32 * 16 * sizeof(_Float16));
  float*    sums   = (float*)carve(64 * sizeof(float));
  float*    gx     = (float*)carve((size_t)G * HDIM * sizeof(float));
  float*    cnt    = (float*)carve((size_t)G * sizeof(float));

  auto cdiv = [](int a, int b) { return (a + b - 1) / b; };
  const int packTot = 33 * 2 * 32 * 16;

  gnn_pack_B<<<cdiv(packTot, 256), 256, 0, stream>>>(nn_W0, nn_b0, packB0);
  gnn_pack_B<<<cdiv(packTot, 256), 256, 0, stream>>>(nn_W1, nn_b1, packB1);
  gnn_node_embed<<<cdiv(N * HDIM, 256), 256, 0, stream>>>(x, node_W, node_b, hbuf, N);
  gnn_edge_embed<<<cdiv(E * HDIM, 256), 256, 0, stream>>>(edge_attr, edge_W, edge_b, ebuf, E);

  const int nTiles = cdiv(E, 16);
  const int wavesNeeded = cdiv(nTiles, TILES_PER_WAVE);
  const int msgBlocks = cdiv(wavesNeeded, 8);   // 8 waves (256 thr) per block

  for (int layer = 0; layer < 2; ++layer) {
    const _Float16* pB  = layer ? packB1 : packB0;
    const float* root   = layer ? root1 : root0;
    const float* cb     = layer ? cb1   : cb0;
    const float* gam    = layer ? g1    : g0;
    const float* bet    = layer ? be1   : be0;
    gnn_zero_f32<<<cdiv(N * HDIM, 256), 256, 0, stream>>>(agg, N * HDIM);
    gnn_zero_f32<<<1, 64, 0, stream>>>(sums, 64);
    gnn_edge_msg_wmma<<<msgBlocks, 256, 0, stream>>>(
        hbuf, ebuf, srcI, dstI, pB, agg, E, nTiles);
    gnn_root_bnstat<<<cdiv(N * HDIM, 256), 256, 0, stream>>>(hbuf, root, cb, agg, sums, N);
    gnn_bn_relu<<<cdiv(N * HDIM, 256), 256, 0, stream>>>(agg, sums, gam, bet, hbuf, N);
  }

  gnn_zero_f32<<<cdiv(G * HDIM, 256), 256, 0, stream>>>(gx, G * HDIM);
  gnn_zero_f32<<<cdiv(G, 256), 256, 0, stream>>>(cnt, G);
  gnn_pool<<<cdiv(N * HDIM, 256), 256, 0, stream>>>(hbuf, batch, gx, cnt, N);
  gnn_head<<<cdiv(G, 128), 128, 0, stream>>>(gx, cnt, lin1_W, lin1_b, lin2_W, lin2_b,
                                             (float*)d_out, G);
}
